// AdvancedGATRecommender_78331613545099
// MI455X (gfx1250) — compile-verified, matrix operand order checked by reading, not run on hardware
//
#include <hip/hip_runtime.h>
#include <hip/hip_bf16.h>
#include <math.h>

// ---------------------------------------------------------------------------
// CDNA5 (gfx1250) GATv2 stack: bf16 WMMA GEMMs + scatter/softmax edge kernels
// ---------------------------------------------------------------------------

#define TPB 256
#define WAVES_PER_BLOCK 8   // 256 threads / wave32
#define KMAX 512            // largest GEMM K in this model (2*eff = 512)

typedef __attribute__((ext_vector_type(16))) __bf16       v16bf;
typedef __attribute__((ext_vector_type(8)))  float        v8f;
typedef __attribute__((ext_vector_type(4)))  unsigned int v4u;

union FragAB { v16bf v; v4u q[2]; };

__device__ __forceinline__ unsigned short f32_to_bf16(float f) {
  unsigned int u = __float_as_uint(f);
  u += 0x7FFFu + ((u >> 16) & 1u);      // round-to-nearest-even
  return (unsigned short)(u >> 16);
}

__device__ __forceinline__ void atomic_max_float(float* addr, float val) {
  // Ordered-bits trick; valid with -FLT_MAX init.
  if (val >= 0.f) atomicMax((int*)addr, __float_as_int(val));
  else            atomicMin((unsigned int*)addr, __float_as_uint(val));
}

// ---------------------------------------------------------------------------
// C[N x M] = act( A[N x K](bf16) @ B[K x M] + bias ),  B given transposed MxK.
// Block tile: 128 rows x 64 cols. 8 waves x (16 rows x 4 col-tiles).
// Full 64 x K bf16 B panel staged in LDS once (<= 64KB); K loop is
// barrier-free: each A fragment (2 x global b128) feeds 4 WMMAs.
// ---------------------------------------------------------------------------
__global__ __launch_bounds__(TPB)
void wmma_gemm_kernel(const unsigned short* __restrict__ A,
                      const unsigned short* __restrict__ Bt,
                      const float* __restrict__ bias,
                      float* __restrict__ C,
                      unsigned short* __restrict__ Cbf,   // optional bf16 mirror
                      int Nrows, int K, int M, int relu)
{
  __shared__ unsigned short Bs[64 * KMAX];   // 64KB max

  const int lane = threadIdx.x & 31;
  const int wave = threadIdx.x >> 5;
  const int half = lane >> 4;          // 0 or 1
  const int l16  = lane & 15;
  const int nRowTiles = (Nrows + 15) >> 4;

  int rowTile = blockIdx.x * WAVES_PER_BLOCK + wave;
  if (rowTile >= nRowTiles) rowTile = nRowTiles - 1;  // keep all waves at barrier

  const int colBase = blockIdx.y << 6;      // 64 columns per block
  const int row = (rowTile << 4) + l16;
  const int k0  = half << 3;                // K run base: 0 or 8 (then +16)

  // ---- stage B panel: 64 cols x K bf16, cooperative b128 loads ----
  {
    const int chunksPerCol = K >> 3;               // uint4 chunks per column
    const int totalChunks  = 64 * chunksPerCol;
    for (int q = threadIdx.x; q < totalChunks; q += TPB) {
      const int c  = q / chunksPerCol;
      const int kp = (q - c * chunksPerCol) << 3;
      *(v4u*)&Bs[c * K + kp] = *(const v4u*)(Bt + (size_t)(colBase + c) * K + kp);
    }
  }
  __syncthreads();

  const unsigned short* arow = A + (size_t)row * K;

  v8f acc[4];
  #pragma unroll
  for (int t = 0; t < 4; ++t) acc[t] = (v8f){0.f,0.f,0.f,0.f,0.f,0.f,0.f,0.f};

  for (int kb = 0; kb < K; kb += 32) {
    __builtin_prefetch((const void*)(arow + kb + 32), 0, 1);

    FragAB a;
    a.q[0] = *(const v4u*)(arow + kb + k0);
    a.q[1] = *(const v4u*)(arow + kb + k0 + 16);

    #pragma unroll
    for (int t = 0; t < 4; ++t) {
      FragAB b;
      const unsigned short* bcol = &Bs[((t << 4) + l16) * K + kb + k0];
      b.q[0] = *(const v4u*)bcol;
      b.q[1] = *(const v4u*)(bcol + 16);
      acc[t] = __builtin_amdgcn_wmma_f32_16x16x32_bf16(false, a.v, false, b.v,
                                                       (short)0, acc[t], false, false);
    }
  }

  const int rbase = (rowTile << 4) + (half << 3);
  #pragma unroll
  for (int t = 0; t < 4; ++t) {
    const int col = colBase + (t << 4) + l16;
    const float bv = bias ? bias[col] : 0.f;
    for (int r = 0; r < 8; ++r) {
      float v = acc[t][r] + bv;
      if (relu) v = fmaxf(v, 0.f);
      size_t idx = (size_t)(rbase + r) * M + col;
      C[idx] = v;
      if (Cbf) Cbf[idx] = f32_to_bf16(v);
    }
  }
}

// ---------------------------------------------------------------------------
// Small utility kernels
// ---------------------------------------------------------------------------
__global__ void f32_to_bf16_kernel(const float* __restrict__ in,
                                   unsigned short* __restrict__ out, int n) {
  int i = blockIdx.x * blockDim.x + threadIdx.x;
  if (i < n) out[i] = f32_to_bf16(in[i]);
}

// W: K x M  ->  Wt: M x K (bf16)
__global__ void transpose_bf16_kernel(const float* __restrict__ W,
                                      unsigned short* __restrict__ Wt,
                                      int K, int M) {
  int i = blockIdx.x * blockDim.x + threadIdx.x;
  if (i >= K * M) return;
  int k = i / M, m = i - k * M;
  Wt[(size_t)m * K + k] = f32_to_bf16(W[i]);
}

__global__ void fill_kernel(float* __restrict__ p, int n, float v) {
  int i = blockIdx.x * blockDim.x + threadIdx.x;
  if (i < n) p[i] = v;
}

// ---------------------------------------------------------------------------
// Edge phase
// ---------------------------------------------------------------------------
// one wave per edge: s = leaky_relu(xl[src]+xr[dst]+ea*We); logits = <s, att>
__global__ __launch_bounds__(TPB)
void edge_logits_max_kernel(const float* __restrict__ xl, const float* __restrict__ xr,
                            const float* __restrict__ We, const float* __restrict__ att,
                            const float* __restrict__ ea,
                            const int* __restrict__ src, const int* __restrict__ dst,
                            float* __restrict__ logits, float* __restrict__ m,
                            int E, int H, int C)
{
  const int lane = threadIdx.x & 31, wave = threadIdx.x >> 5;
  const int e = blockIdx.x * WAVES_PER_BLOCK + wave;
  if (e >= E) return;
  const int HC = H * C;
  const int s = src[e], d = dst[e];
  const float eav = ea[e];
  const int per = HC >> 5;            // channels per lane (8 or 2)
  const int cb = lane * per;
  float part = 0.f;
  for (int i = 0; i < per; ++i) {
    int ch = cb + i;
    float v = xl[(size_t)s * HC + ch] + xr[(size_t)d * HC + ch] + eav * We[ch];
    v = v > 0.f ? v : 0.2f * v;
    part += v * att[ch];
  }
  const int lph = C / per;            // lanes per head (8 or 32), aligned groups
  for (int off = lph >> 1; off; off >>= 1) part += __shfl_xor(part, off, 32);
  if ((lane & (lph - 1)) == 0) {
    int hh = cb / C;
    logits[(size_t)e * H + hh] = part;
    atomic_max_float(&m[(size_t)d * H + hh], part);
  }
}

__global__ void edge_exp_kernel(const float* __restrict__ logits,
                                const float* __restrict__ m,
                                const int* __restrict__ dst,
                                float* __restrict__ ex, float* __restrict__ den,
                                int E, int H)
{
  int idx = blockIdx.x * blockDim.x + threadIdx.x;
  if (idx >= E * H) return;
  int e = idx / H, hh = idx - e * H;
  int d = dst[e];
  float v = expf(logits[idx] - m[(size_t)d * H + hh]);
  ex[idx] = v;
  atomicAdd(&den[(size_t)d * H + hh], v);
}

__global__ void edge_agg_kernel(const float* __restrict__ xl,
                                const float* __restrict__ ex, const float* __restrict__ den,
                                const int* __restrict__ src, const int* __restrict__ dst,
                                float* __restrict__ agg, int total, int H, int C)
{
  int idx = blockIdx.x * blockDim.x + threadIdx.x;
  if (idx >= total) return;
  const int HC = H * C;
  int e = idx / HC, ch = idx - e * HC, hh = ch / C;
  int d = dst[e];
  float alpha = ex[(size_t)e * H + hh] / (den[(size_t)d * H + hh] + 1e-16f);
  atomicAdd(&agg[(size_t)d * HC + ch], xl[(size_t)src[e] * HC + ch] * alpha);
}

// ---------------------------------------------------------------------------
// Node epilogues (one wave per node)
// ---------------------------------------------------------------------------
__global__ __launch_bounds__(TPB)
void post_kernel(const float* __restrict__ agg, const float* __restrict__ bias,
                 const float* __restrict__ Wg, const float* __restrict__ bg,
                 const float* __restrict__ xp,
                 float* __restrict__ h, unsigned short* __restrict__ hbf,
                 int N, int H, int C, int eff, int concat)
{
  const int lane = threadIdx.x & 31, wave = threadIdx.x >> 5;
  const int n = blockIdx.x * WAVES_PER_BLOCK + wave;
  if (n >= N) return;
  const int HC = H * C;
  const int per = eff >> 5;           // 8 or 2
  float hv[8];
  float gdot = 0.f;
  for (int i = 0; i < per; ++i) {
    int c = lane * per + i;
    float o;
    if (concat) o = agg[(size_t)n * HC + c];
    else { o = 0.f; for (int hh = 0; hh < H; ++hh) o += agg[(size_t)n * HC + hh * C + c];
           o /= (float)H; }
    o += bias[c];
    float e = o > 0.f ? o : (expf(o) - 1.f);   // ELU
    hv[i] = e;
    gdot += e * Wg[c];
  }
  for (int off = 16; off; off >>= 1) gdot += __shfl_xor(gdot, off, 32);
  float g = 1.f / (1.f + expf(-(gdot + bg[0])));
  for (int i = 0; i < per; ++i) {
    int c = lane * per + i;
    float v = g * hv[i] + (1.f - g) * xp[(size_t)n * eff + c];
    h[(size_t)n * eff + c] = v;
    hbf[(size_t)n * eff + c] = f32_to_bf16(v);
  }
}

__global__ __launch_bounds__(TPB)
void layernorm_kernel(const float* __restrict__ h, const float* __restrict__ f,
                      const float* __restrict__ gamma, const float* __restrict__ beta,
                      float* __restrict__ out, int N, int eff)
{
  const int lane = threadIdx.x & 31, wave = threadIdx.x >> 5;
  const int n = blockIdx.x * WAVES_PER_BLOCK + wave;
  if (n >= N) return;
  const int per = eff >> 5;
  float v[8], s = 0.f, s2 = 0.f;
  for (int i = 0; i < per; ++i) {
    int c = lane * per + i;
    float x = h[(size_t)n * eff + c] + f[(size_t)n * eff + c];
    v[i] = x; s += x; s2 += x * x;
  }
  for (int off = 16; off; off >>= 1) { s += __shfl_xor(s, off, 32); s2 += __shfl_xor(s2, off, 32); }
  float mu = s / (float)eff;
  float var = s2 / (float)eff - mu * mu;
  float rs = rsqrtf(var + 1e-5f);
  for (int i = 0; i < per; ++i) {
    int c = lane * per + i;
    out[(size_t)n * eff + c] = (v[i] - mu) * rs * gamma[c] + beta[c];
  }
}

// ---------------------------------------------------------------------------
// Host orchestration
// ---------------------------------------------------------------------------
static const int L_din[3]  = {64, 256, 256};
static const int L_H[3]    = {4, 4, 1};
static const int L_C[3]    = {64, 64, 64};
static const int L_proj[3] = {1, 0, 1};

static inline int cdiv(int a, int b) { return (a + b - 1) / b; }

extern "C" void kernel_launch(void* const* d_in, const int* in_sizes, int n_in,
                              void* d_out, int out_size, void* d_ws, size_t ws_size,
                              hipStream_t stream)
{
  (void)in_sizes; (void)n_in; (void)out_size; (void)ws_size;
  const int N = 50000, E = 400000;
  const float* node_emb   = (const float*)d_in[0];
  const float* edge_attr  = (const float*)d_in[1];
  const int*   edge_index = (const int*)d_in[2];
  const int* src = edge_index;
  const int* dst = edge_index + E;

  // Params per layer in dict order:
  // 0 Wl 1 bl 2 Wr 3 br 4 We 5 att 6 bias 7 Wg 8 bg 9 W1 10 b1 11 W2 12 b2
  // 13 gamma 14 beta [15 Wp 16 bp]
  const float* P[3][17];
  {
    int idx = 3;
    for (int l = 0; l < 3; ++l) {
      for (int j = 0; j < 15; ++j) P[l][j] = (const float*)d_in[idx++];
      if (L_proj[l]) { P[l][15] = (const float*)d_in[idx++];
                       P[l][16] = (const float*)d_in[idx++]; }
      else           { P[l][15] = nullptr; P[l][16] = nullptr; }
    }
  }

  char* wp = (char*)d_ws;
  auto carve = [&](size_t bytes) -> void* {
    void* r = (void*)wp;
    wp += (bytes + 255) & ~(size_t)255;
    return r;
  };

  unsigned short* xbf = (unsigned short*)carve((size_t)N * 256 * 2);
  unsigned short* wWl = (unsigned short*)carve(256 * 256 * 2);
  unsigned short* wWr = (unsigned short*)carve(256 * 256 * 2);
  unsigned short* wWp = (unsigned short*)carve(256 * 256 * 2);
  unsigned short* wW1 = (unsigned short*)carve(512 * 256 * 2);
  unsigned short* wW2 = (unsigned short*)carve(512 * 256 * 2);
  float* xl   = (float*)carve((size_t)N * 256 * 4);
  float* xr   = (float*)carve((size_t)N * 256 * 4);
  float* xpB  = (float*)carve((size_t)N * 256 * 4);
  float* hB   = (float*)carve((size_t)N * 256 * 4);
  float* fB   = (float*)carve((size_t)N * 256 * 4);
  float* tB   = (float*)carve((size_t)N * 512 * 4);
  unsigned short* hbf = (unsigned short*)carve((size_t)N * 256 * 2);
  unsigned short* tbf = (unsigned short*)carve((size_t)N * 512 * 2);
  float* logits = (float*)carve((size_t)E * 4 * 4);
  float* exB    = (float*)carve((size_t)E * 4 * 4);
  float* mB     = (float*)carve((size_t)N * 4 * 4);
  float* denB   = (float*)carve((size_t)N * 4 * 4);
  float* agg    = (float*)carve((size_t)N * 256 * 4);
  float* bufX   = (float*)carve((size_t)N * 256 * 4);
  float* bufY   = (float*)carve((size_t)N * 256 * 4);

  const float* xin = node_emb;
  float* louts[3] = { bufX, bufY, (float*)d_out };
  const dim3 blk(TPB);
  const int nRowTiles = cdiv(N, 16);     // 3125, exact
  const int gx = cdiv(nRowTiles, WAVES_PER_BLOCK);

  for (int l = 0; l < 3; ++l) {
    const int din = L_din[l], H = L_H[l], C = L_C[l];
    const int HC = H * C;
    const int concat = (H > 1);
    const int eff = concat ? HC : C;

    f32_to_bf16_kernel<<<cdiv(N * din, TPB), blk, 0, stream>>>(xin, xbf, N * din);

    transpose_bf16_kernel<<<cdiv(din * HC, TPB), blk, 0, stream>>>(P[l][0], wWl, din, HC);
    transpose_bf16_kernel<<<cdiv(din * HC, TPB), blk, 0, stream>>>(P[l][2], wWr, din, HC);
    if (L_proj[l])
      transpose_bf16_kernel<<<cdiv(din * eff, TPB), blk, 0, stream>>>(P[l][15], wWp, din, eff);
    transpose_bf16_kernel<<<cdiv(eff * 2 * eff, TPB), blk, 0, stream>>>(P[l][9],  wW1, eff, 2 * eff);
    transpose_bf16_kernel<<<cdiv(2 * eff * eff, TPB), blk, 0, stream>>>(P[l][11], wW2, 2 * eff, eff);

    dim3 gL(gx, HC / 64);
    wmma_gemm_kernel<<<gL, blk, 0, stream>>>(xbf, wWl, P[l][1], xl,
                                             (unsigned short*)nullptr, N, din, HC, 0);
    wmma_gemm_kernel<<<gL, blk, 0, stream>>>(xbf, wWr, P[l][3], xr,
                                             (unsigned short*)nullptr, N, din, HC, 0);
    const float* xpPtr = xin;            // no-proj layers have din == eff
    if (L_proj[l]) {
      dim3 gP(gx, eff / 64);
      wmma_gemm_kernel<<<gP, blk, 0, stream>>>(xbf, wWp, P[l][16], xpB,
                                               (unsigned short*)nullptr, N, din, eff, 0);
      xpPtr = xpB;
    }

    fill_kernel<<<cdiv(N * H, TPB), blk, 0, stream>>>(mB, N * H, -3.402823466e38f);
    fill_kernel<<<cdiv(N * H, TPB), blk, 0, stream>>>(denB, N * H, 0.f);
    fill_kernel<<<cdiv(N * HC, TPB), blk, 0, stream>>>(agg, N * HC, 0.f);

    edge_logits_max_kernel<<<cdiv(E, WAVES_PER_BLOCK), blk, 0, stream>>>(
        xl, xr, P[l][4], P[l][5], edge_attr, src, dst, logits, mB, E, H, C);
    edge_exp_kernel<<<cdiv(E * H, TPB), blk, 0, stream>>>(logits, mB, dst, exB, denB, E, H);
    edge_agg_kernel<<<cdiv(E * HC, TPB), blk, 0, stream>>>(
        xl, exB, denB, src, dst, agg, E * HC, H, C);

    post_kernel<<<cdiv(N, WAVES_PER_BLOCK), blk, 0, stream>>>(
        agg, P[l][6], P[l][7], P[l][8], xpPtr, hB, hbf, N, H, C, eff, concat);

    dim3 gF1(gx, (2 * eff) / 64);
    wmma_gemm_kernel<<<gF1, blk, 0, stream>>>(hbf, wW1, P[l][10], tB, tbf, N, eff, 2 * eff, 1);
    dim3 gF2(gx, eff / 64);
    wmma_gemm_kernel<<<gF2, blk, 0, stream>>>(tbf, wW2, P[l][12], fB,
                                              (unsigned short*)nullptr, N, 2 * eff, eff, 0);

    layernorm_kernel<<<cdiv(N, WAVES_PER_BLOCK), blk, 0, stream>>>(
        hB, fB, P[l][13], P[l][14], louts[l], N, eff);

    xin = louts[l];
  }
}
